// GRUModel_79955111182519
// MI455X (gfx1250) — matrix operand order repaired and checked by reading, not compiled, not run on hardware
//
#include <hip/hip_runtime.h>
#include <math.h>

typedef __attribute__((ext_vector_type(16))) __bf16 v16bf;
typedef __attribute__((ext_vector_type(8)))  float  v8f;

#define HID   250
#define GATES 750
#define BATCH 128
#define TLEN  1024
#define IN    300
#define IPAD  320
#define HPAD  256
#define GPAD  768
#define M_TOT (BATCH*TLEN)   /* 131072 */

union BOp { int4 i[2]; v16bf v; };

/* ---------------- kernel 0: pack weights to bf16, padded, [n][k] k-contig ---------------- */
__global__ void __launch_bounds__(256)
pack_weights(const float* __restrict__ Wih, const float* __restrict__ Whh,
             __bf16* __restrict__ wih_p, __bf16* __restrict__ whh_p) {
    int tid = blockIdx.x * blockDim.x + threadIdx.x;
    const int n1 = GPAD * IPAD;
    const int n2 = GPAD * HPAD;
    if (tid < n1) {
        int n = tid / IPAD, k = tid % IPAD;
        float v = (n < GATES && k < IN) ? Wih[n * IN + k] : 0.0f;
        wih_p[tid] = (__bf16)v;
    } else if (tid < n1 + n2) {
        int t = tid - n1;
        int n = t / HPAD, k = t % HPAD;
        float v = (n < GATES && k < HID) ? Whh[n * HID + k] : 0.0f;
        whh_p[t] = (__bf16)v;
    }
}

/* ---------------- kernel 1: xi = x @ Wih^T + b_ih  (M=131072, N=768, K=320) -------------
   Block = 256 threads = 8 waves. Block tile: 128(M) x 64(N). Each wave: 16(M) x 64(N).   */
#define LSTR 40   /* LDS row stride in bf16 (80B, 16B-aligned, bank-skewed) */
__global__ void __launch_bounds__(256)
input_gemm(const float* __restrict__ x, const __bf16* __restrict__ wih_p,
           const float* __restrict__ b_ih, float* __restrict__ xi) {
    __shared__ __bf16 lA[128 * LSTR];
    __shared__ __bf16 lB[64 * LSTR];
    const int tid  = threadIdx.x;
    const int wave = tid >> 5, lane = tid & 31;
    const int grp  = lane >> 4, lan16 = lane & 15;
    const long mbase = (long)blockIdx.y * 128;
    const int  nbase = blockIdx.x * 64;

    v8f acc[4] = {};
    for (int kc = 0; kc < IPAD; kc += 32) {
        __syncthreads();
        { /* A tile: 128 rows x 32 k (f32 -> bf16), 2 threads/row */
            int r = tid >> 1, half = tid & 1;
            int kb = kc + half * 16;
            const float* src = x + (mbase + r) * IN + kb;
            __bf16* dst = lA + r * LSTR + half * 16;
            #pragma unroll
            for (int q = 0; q < 16; ++q) {
                int k = kb + q;
                dst[q] = (__bf16)((k < IN) ? src[q] : 0.0f);
            }
        }
        { /* B tile: 64 n x 32 k, already packed bf16, 4 threads/row */
            int r = tid >> 2, part = tid & 3;
            const __bf16* src = wih_p + (nbase + r) * IPAD + kc + part * 8;
            __bf16* dst = lB + r * LSTR + part * 8;
            #pragma unroll
            for (int q = 0; q < 8; ++q) dst[q] = src[q];
        }
        __syncthreads();

        BOp a;  /* 16-bit A layout: lane -> row lan16, K = grp*16 .. +15 contiguous */
        const __bf16* arow = lA + (wave * 16 + lan16) * LSTR;
        a.i[0] = *(const int4*)(arow + grp * 16);
        a.i[1] = *(const int4*)(arow + grp * 16 + 8);
        #pragma unroll
        for (int nt = 0; nt < 4; ++nt) {
            BOp b; /* B layout: lane -> col lan16, K = grp*16 .. +15 contiguous */
            const __bf16* brow = lB + (nt * 16 + lan16) * LSTR + grp * 16;
            b.i[0] = *(const int4*)(brow);
            b.i[1] = *(const int4*)(brow + 8);
            acc[nt] = __builtin_amdgcn_wmma_f32_16x16x32_bf16(
                false, a.v, false, b.v, (short)0, acc[nt], false, false);
        }
    }
    /* epilogue: +bias, store f32 xi (padded stride GPAD). C layout: VGPR r -> M=r+8*grp, N=lan16 */
    #pragma unroll
    for (int nt = 0; nt < 4; ++nt) {
        int n = nbase + nt * 16 + lan16;
        float bias = (n < GATES) ? b_ih[n] : 0.0f;
        #pragma unroll
        for (int r = 0; r < 8; ++r) {
            long m = mbase + wave * 16 + r + grp * 8;
            xi[m * GPAD + n] = acc[nt][r] + bias;
        }
    }
}

/* ---------------- kernel 2: recurrent scan. 8 blocks (16 batch rows each), 512 thr = 16 waves.
   W_hh k-tiles 0..3 live in VGPRs across the whole scan; k-tiles 4..7 stream from L2.
   h state lives in per-thread registers (f32) + LDS (bf16 WMMA A operand). Biases in regs. */
__global__ void __launch_bounds__(512)
gru_scan(const float* __restrict__ xi, const __bf16* __restrict__ whh_p,
         const float* __restrict__ b_hh, float* __restrict__ out) {
    __shared__ float  gh[16 * GPAD];   /* 48 KB */
    __shared__ __bf16 hb[16 * 264];    /* h in bf16, padded K, 528B rows (16B aligned) */
    const int tid  = threadIdx.x;
    const int wave = tid >> 5, lane = tid & 31;
    const int grp  = lane >> 4, lan16 = lane & 15;
    const int bb   = blockIdx.x * 16;
    const int jrow = wave;             /* gate loop: one h-row per wave, lanes stride cols */

    for (int e = tid; e < 16 * 264; e += 512) hb[e] = (__bf16)0.0f;

    /* t-invariant per-thread state: h registers + gate biases */
    float hreg[8], bh0[8], bh1[8], bh2[8];
    #pragma unroll
    for (int q = 0; q < 8; ++q) {
        int h = lane + 32 * q;
        bool ok = h < HID;
        hreg[q] = 0.0f;
        bh0[q] = ok ? b_hh[h] : 0.0f;
        bh1[q] = ok ? b_hh[h + HID] : 0.0f;
        bh2[q] = ok ? b_hh[h + 2 * HID] : 0.0f;
    }

    /* hoist W_hh k-tiles 0..3 for this wave's 3 n-tiles into registers (96 VGPRs) */
    BOp bpre[3][4];
    #pragma unroll
    for (int i = 0; i < 3; ++i) {
        int nt = wave * 3 + i;
        const __bf16* bbase = whh_p + (nt * 16 + lan16) * HPAD + grp * 16;
        #pragma unroll
        for (int kt = 0; kt < 4; ++kt) {
            bpre[i][kt].i[0] = *(const int4*)(bbase + kt * 32);
            bpre[i][kt].i[1] = *(const int4*)(bbase + kt * 32 + 8);
        }
    }
    __syncthreads();

    for (int t = 0; t < TLEN; ++t) {
        /* A operands: all 8 k-tiles of h, shared by every n-tile this wave computes */
        BOp a[8];
        const __bf16* arow = hb + lan16 * 264;
        #pragma unroll
        for (int kt = 0; kt < 8; ++kt) {
            a[kt].i[0] = *(const int4*)(arow + kt * 32 + grp * 16);
            a[kt].i[1] = *(const int4*)(arow + kt * 32 + grp * 16 + 8);
        }
        #pragma unroll
        for (int i = 0; i < 3; ++i) {
            int nt = wave * 3 + i;        /* 16 waves x 3 = 48 n-tiles = 768 cols */
            v8f acc = {};
            #pragma unroll
            for (int kt = 0; kt < 4; ++kt)   /* register-resident half of W_hh */
                acc = __builtin_amdgcn_wmma_f32_16x16x32_bf16(
                    false, a[kt].v, false, bpre[i][kt].v, (short)0, acc, false, false);
            const __bf16* bbase = whh_p + (nt * 16 + lan16) * HPAD + grp * 16;
            #pragma unroll
            for (int kt = 4; kt < 8; ++kt) { /* streamed (L2-resident) half */
                BOp b;
                b.i[0] = *(const int4*)(bbase + kt * 32);
                b.i[1] = *(const int4*)(bbase + kt * 32 + 8);
                acc = __builtin_amdgcn_wmma_f32_16x16x32_bf16(
                    false, a[kt].v, false, b.v, (short)0, acc, false, false);
            }
            int n = nt * 16 + lan16;
            #pragma unroll
            for (int r = 0; r < 8; ++r)
                gh[(r + grp * 8) * GPAD + n] = acc[r];
        }
        __syncthreads();

        /* gates + h update: row j = wave, cols h = lane + 32q */
        {
            const long row = (long)(bb + jrow) * TLEN + t;
            const float* xir = xi + row * GPAD;
            /* one cacheline-granular prefetch per thread covers next step's whole xi row:
               32 lanes x 96B stride spans 3072B = GPAD floats */
            if (t + 1 < TLEN) __builtin_prefetch(xir + GPAD + lane * 24, 0, 0);
            #pragma unroll
            for (int q = 0; q < 8; ++q) {
                int h = lane + 32 * q;
                if (h < HID) {
                    float xr = xir[h], xz = xir[h + HID], xn = xir[h + 2 * HID];
                    float hr = gh[jrow * GPAD + h]            + bh0[q];
                    float hz = gh[jrow * GPAD + h + HID]      + bh1[q];
                    float hn = gh[jrow * GPAD + h + 2 * HID]  + bh2[q];
                    float r = 1.0f / (1.0f + __expf(-(xr + hr)));
                    float z = 1.0f / (1.0f + __expf(-(xz + hz)));
                    float n = tanhf(xn + r * hn);
                    float hnew = (1.0f - z) * n + z * hreg[q];
                    hreg[q] = hnew;
                    hb[jrow * 264 + h] = (__bf16)hnew;
                    out[row * HID + h] = hnew;
                }
            }
        }
        __syncthreads();
    }
}

/* ---------------- kernel 3: softmax over time axis, in place on d_out ---------------- */
__global__ void __launch_bounds__(256)
softmax_time(float* __restrict__ out) {
    int b = blockIdx.x;
    int h = threadIdx.x;
    if (h >= HID) return;
    float* base = out + (long)b * TLEN * HID + h;
    float m = -INFINITY;
    for (int t = 0; t < TLEN; ++t) m = fmaxf(m, base[(long)t * HID]);
    float s = 0.0f;
    for (int t = 0; t < TLEN; ++t) s += __expf(base[(long)t * HID] - m);
    float inv = 1.0f / s;
    for (int t = 0; t < TLEN; ++t) base[(long)t * HID] = __expf(base[(long)t * HID] - m) * inv;
}

extern "C" void kernel_launch(void* const* d_in, const int* in_sizes, int n_in,
                              void* d_out, int out_size, void* d_ws, size_t ws_size,
                              hipStream_t stream) {
    const float* x   = (const float*)d_in[0];
    const float* Wih = (const float*)d_in[1];
    const float* Whh = (const float*)d_in[2];
    const float* bih = (const float*)d_in[3];
    const float* bhh = (const float*)d_in[4];
    float* out = (float*)d_out;

    /* workspace layout: [0, 884736) packed weights (bf16), [1MB, 1MB+402.7MB) xi f32 */
    __bf16* wih_p = (__bf16*)d_ws;
    __bf16* whh_p = wih_p + (size_t)GPAD * IPAD;
    float*  xi    = (float*)((char*)d_ws + (1u << 20));

    int packN = GPAD * IPAD + GPAD * HPAD;
    hipLaunchKernelGGL(pack_weights, dim3((packN + 255) / 256), dim3(256), 0, stream,
                       Wih, Whh, wih_p, whh_p);
    hipLaunchKernelGGL(input_gemm, dim3(GPAD / 64, M_TOT / 128), dim3(256), 0, stream,
                       x, wih_p, bih, xi);
    hipLaunchKernelGGL(gru_scan, dim3(BATCH / 16), dim3(512), 0, stream,
                       xi, whh_p, bhh, out);
    hipLaunchKernelGGL(softmax_time, dim3(BATCH), dim3(256), 0, stream, out);
}
